// LeanContext1D_9517647528269
// MI455X (gfx1250) — compile-verified, matrix operand order checked by reading, not run on hardware
//
#include <hip/hip_runtime.h>
#include <hip/hip_bf16.h>

// ---------------------------------------------------------------------------
// LeanContext1D on MI455X (gfx1250): bf16 WMMA GEMMs + flash attention,
// async global->LDS double-buffered tiles.
// ---------------------------------------------------------------------------

#define B_  2
#define T_  2048
#define C_  1024
#define H_  16
#define D_  64
#define P_  1280
#define M_  (B_*T_)          // 4096 rows

typedef __attribute__((ext_vector_type(16))) __bf16 v16bf;
typedef __attribute__((ext_vector_type(8)))  float  v8f;

union Frag16 { uint4 u[2]; v16bf v; };

static __device__ inline v8f vzero8() {
    v8f z;
#pragma unroll
    for (int i = 0; i < 8; ++i) z[i] = 0.0f;
    return z;
}

// Async copy 16 bytes global -> LDS (tracked by ASYNCcnt).
static __device__ inline void async_copy16(const void* gptr, void* lptr) {
    uint64_t ga = (uint64_t)(uintptr_t)gptr;
    uint32_t la = (uint32_t)(uintptr_t)lptr;   // low 32 bits = LDS offset
    asm volatile("global_load_async_to_lds_b128 %0, %1, off"
                 :: "v"(la), "v"(ga) : "memory");
}
static __device__ inline void wait_async0() {
    asm volatile("s_wait_asynccnt 0" ::: "memory");
}

// ---------------------------------------------------------------------------
// Convert fp32 -> bf16 (linear)
// ---------------------------------------------------------------------------
__global__ void cvt_bf16_kernel(const float* __restrict__ in, __bf16* __restrict__ out, int n) {
    int i = blockIdx.x * 256 + threadIdx.x;
    if (i < n) out[i] = (__bf16)in[i];
}

// Transpose + convert: in [rows][cols] fp32 -> out [cols][rows] bf16
__global__ void transpose_cvt_kernel(const float* __restrict__ in, __bf16* __restrict__ out,
                                     int rows, int cols) {
    int i = blockIdx.x * 256 + threadIdx.x;
    int n = rows * cols;
    if (i < n) {
        int r = i / cols, c = i % cols;
        out[(size_t)c * rows + r] = (__bf16)in[i];
    }
}

// ---------------------------------------------------------------------------
// bf16 WMMA GEMM:  Y[m][n] = sum_c A[m][c] * WT[n][c] + bias[n]
// A: [M_ x C_] bf16 row-major.  WT: [N x C_] bf16 (pre-transposed weight).
// mode 0: q  -> [B,H,T,D] bf16, scaled by 1/sqrt(D)
// mode 1: k  -> [B,H,T,D] bf16
// mode 2: v  -> [B,H,D,T] bf16 (transposed for PV B-fragments)
// mode 3: h  -> [M_ x P_] fp32
// ---------------------------------------------------------------------------
__global__ __launch_bounds__(256)
void gemm_bf16_kernel(const __bf16* __restrict__ Abf, const __bf16* __restrict__ WT,
                      const float* __restrict__ bias, void* __restrict__ Out,
                      int mode) {
    __shared__ __bf16 sA[2][128 * 40];   // double-buffered 128x32 tiles, stride 40 halves
    __shared__ __bf16 sB[2][128 * 40];

    const int tid  = threadIdx.x;
    const int wave = tid >> 5;
    const int lane = tid & 31;
    const int l16  = lane & 15;
    const int half = lane >> 4;
    const int m0   = blockIdx.x * 128;
    const int n0   = blockIdx.y * 128;

    // Per-thread fill assignment: chunks tid and tid+256 (row = c>>2, seg = c&3).
    const int rowA = tid >> 2,           segA = tid & 3;
    const int rowB = (tid + 256) >> 2,   segB = (tid + 256) & 3;

    v8f acc[8];
#pragma unroll
    for (int i = 0; i < 8; ++i) acc[i] = vzero8();

    // Prefetch first K-chunk into buffer 0.
    {
        const int k0 = 0;
        async_copy16(Abf + (size_t)(m0 + rowA) * C_ + k0 + segA * 8, &sA[0][rowA * 40 + segA * 8]);
        async_copy16(Abf + (size_t)(m0 + rowB) * C_ + k0 + segB * 8, &sA[0][rowB * 40 + segB * 8]);
        async_copy16(WT  + (size_t)(n0 + rowA) * C_ + k0 + segA * 8, &sB[0][rowA * 40 + segA * 8]);
        async_copy16(WT  + (size_t)(n0 + rowB) * C_ + k0 + segB * 8, &sB[0][rowB * 40 + segB * 8]);
    }
    wait_async0();
    __syncthreads();

    for (int it = 0; it < C_ / 32; ++it) {
        const int cur = it & 1;
        // Prefetch next chunk into the other buffer (overlaps with compute below).
        if (it + 1 < C_ / 32) {
            const int nxt = (it + 1) & 1;
            const int k1 = (it + 1) * 32;
            async_copy16(Abf + (size_t)(m0 + rowA) * C_ + k1 + segA * 8, &sA[nxt][rowA * 40 + segA * 8]);
            async_copy16(Abf + (size_t)(m0 + rowB) * C_ + k1 + segB * 8, &sA[nxt][rowB * 40 + segB * 8]);
            async_copy16(WT  + (size_t)(n0 + rowA) * C_ + k1 + segA * 8, &sB[nxt][rowA * 40 + segA * 8]);
            async_copy16(WT  + (size_t)(n0 + rowB) * C_ + k1 + segB * 8, &sB[nxt][rowB * 40 + segB * 8]);
        }

        // A fragment for this wave's 16-row strip.
        Frag16 a;
        {
            int m = wave * 16 + l16;
            a.u[0] = *(const uint4*)(&sA[cur][m * 40 + half * 8]);
            a.u[1] = *(const uint4*)(&sA[cur][m * 40 + 16 + half * 8]);
        }
        // Batch-load all 8 B fragments, then issue 8 WMMAs back-to-back.
        Frag16 bfr[8];
#pragma unroll
        for (int nf = 0; nf < 8; ++nf) {
            int n = nf * 16 + l16;
            bfr[nf].u[0] = *(const uint4*)(&sB[cur][n * 40 + half * 8]);
            bfr[nf].u[1] = *(const uint4*)(&sB[cur][n * 40 + 16 + half * 8]);
        }
#pragma unroll
        for (int nf = 0; nf < 8; ++nf) {
            acc[nf] = __builtin_amdgcn_wmma_f32_16x16x32_bf16(
                false, a.v, false, bfr[nf].v, (short)0, acc[nf], false, false);
        }

        wait_async0();       // prefetched chunk has had the whole compute phase to land
        __syncthreads();
    }

    // Epilogue: scatter per C/D layout (lane = col, VGPR r = row r + 8*half).
#pragma unroll
    for (int nf = 0; nf < 8; ++nf) {
        int n = n0 + nf * 16 + l16;
        float bv = bias[n];
#pragma unroll
        for (int r = 0; r < 8; ++r) {
            int m = m0 + wave * 16 + r + half * 8;
            float val = acc[nf][r] + bv;
            if (mode == 3) {
                ((float*)Out)[(size_t)m * P_ + n] = val;
            } else {
                int b = m / T_, t = m % T_;
                int h = n >> 6, d = n & 63;
                __bf16* o = (__bf16*)Out;
                if (mode == 2) {
                    o[((size_t)((b * H_ + h) * D_ + d)) * T_ + t] = (__bf16)val;
                } else {
                    if (mode == 0) val *= 0.125f;   // 1/sqrt(64)
                    o[((size_t)((b * H_ + h) * T_ + t)) * D_ + d] = (__bf16)val;
                }
            }
        }
    }
}

// ---------------------------------------------------------------------------
// Flash attention.  q,k: [B,H,T,D] bf16 (q pre-scaled).  vT: [B,H,D,T] bf16.
// out: [B,T,C] fp32.  One workgroup = 128 query rows of one (b,h).
// ---------------------------------------------------------------------------
__global__ __launch_bounds__(256)
void attn_kernel(const __bf16* __restrict__ qbf, const __bf16* __restrict__ kbf,
                 const __bf16* __restrict__ vT, float* __restrict__ out) {
    __shared__ __bf16 sP[8 * 16 * 40];   // per-wave 16x32 prob tile, stride 40 halves

    const int tid  = threadIdx.x;
    const int wave = tid >> 5;
    const int lane = tid & 31;
    const int l16  = lane & 15;
    const int half = lane >> 4;
    const int bh   = blockIdx.x;             // 0..B*H-1
    const int b    = bh >> 4, h = bh & 15;
    const int q0   = blockIdx.y * 128 + wave * 16;
    const size_t qkbase = (size_t)bh * T_ * D_;

    // Preload q A-fragments for the two K(d) chunks.
    Frag16 qa0, qa1;
    {
        const __bf16* qrow = qbf + qkbase + (size_t)(q0 + l16) * D_;
        qa0.u[0] = *(const uint4*)(qrow + half * 8);
        qa0.u[1] = *(const uint4*)(qrow + 16 + half * 8);
        qa1.u[0] = *(const uint4*)(qrow + 32 + half * 8);
        qa1.u[1] = *(const uint4*)(qrow + 32 + 16 + half * 8);
    }

    v8f acc[4];
#pragma unroll
    for (int i = 0; i < 4; ++i) acc[i] = vzero8();
    float mrow[8], lrow[8];
#pragma unroll
    for (int r = 0; r < 8; ++r) { mrow[r] = -1e30f; lrow[r] = 0.0f; }

    __bf16* myP = sP + wave * 16 * 40;

    for (int s0 = 0; s0 < T_; s0 += 32) {
        // ---- scores: batch-load 4 k fragments, then 4 WMMAs ----
        Frag16 kb[2][2];
#pragma unroll
        for (int j = 0; j < 2; ++j) {
            const __bf16* krow = kbf + qkbase + (size_t)(s0 + j * 16 + l16) * D_;
            kb[j][0].u[0] = *(const uint4*)(krow + half * 8);
            kb[j][0].u[1] = *(const uint4*)(krow + 16 + half * 8);
            kb[j][1].u[0] = *(const uint4*)(krow + 32 + half * 8);
            kb[j][1].u[1] = *(const uint4*)(krow + 48 + half * 8);
        }
        v8f S[2];
#pragma unroll
        for (int j = 0; j < 2; ++j) {
            v8f z = vzero8();
            z = __builtin_amdgcn_wmma_f32_16x16x32_bf16(false, qa0.v, false, kb[j][0].v, (short)0, z, false, false);
            z = __builtin_amdgcn_wmma_f32_16x16x32_bf16(false, qa1.v, false, kb[j][1].v, (short)0, z, false, false);
            S[j] = z;
        }
        // ---- online softmax: row max (reduce across 16-lane half) ----
        float mnew[8];
#pragma unroll
        for (int r = 0; r < 8; ++r) {
            float v = fmaxf(S[0][r], S[1][r]);
#pragma unroll
            for (int off = 1; off < 16; off <<= 1) v = fmaxf(v, __shfl_xor(v, off, 32));
            mnew[r] = fmaxf(mrow[r], v);
        }
        // ---- probabilities -> LDS (re-layout C-frag -> A-frag), row sums ----
        float rsum[8];
#pragma unroll
        for (int r = 0; r < 8; ++r) {
            float p0 = __expf(S[0][r] - mnew[r]);
            float p1 = __expf(S[1][r] - mnew[r]);
            myP[(r + half * 8) * 40 + l16]      = (__bf16)p0;
            myP[(r + half * 8) * 40 + 16 + l16] = (__bf16)p1;
            float rs = p0 + p1;
#pragma unroll
            for (int off = 1; off < 16; off <<= 1) rs += __shfl_xor(rs, off, 32);
            rsum[r] = rs;
        }
        // ---- rescale running state ----
#pragma unroll
        for (int r = 0; r < 8; ++r) {
            float sc = __expf(mrow[r] - mnew[r]);
            lrow[r] = lrow[r] * sc + rsum[r];
            mrow[r] = mnew[r];
#pragma unroll
            for (int f = 0; f < 4; ++f) acc[f][r] *= sc;
        }
        // ---- P A-fragment from LDS (same-wave DS ops are in-order) ----
        Frag16 pa;
        pa.u[0] = *(const uint4*)(myP + l16 * 40 + half * 8);
        pa.u[1] = *(const uint4*)(myP + l16 * 40 + 16 + half * 8);
        // ---- PV: batch-load 4 V fragments from vT, then 4 WMMAs ----
        Frag16 vb[4];
#pragma unroll
        for (int df = 0; df < 4; ++df) {
            const __bf16* vrow = vT + (size_t)(bh * D_ + df * 16 + l16) * T_ + s0;
            vb[df].u[0] = *(const uint4*)(vrow + half * 8);
            vb[df].u[1] = *(const uint4*)(vrow + 16 + half * 8);
        }
#pragma unroll
        for (int df = 0; df < 4; ++df) {
            acc[df] = __builtin_amdgcn_wmma_f32_16x16x32_bf16(
                false, pa.v, false, vb[df].v, (short)0, acc[df], false, false);
        }
    }

    // ---- normalize + write [b, t, h*64+d] ----
#pragma unroll
    for (int df = 0; df < 4; ++df) {
        int d = df * 16 + l16;
#pragma unroll
        for (int r = 0; r < 8; ++r) {
            int t = q0 + r + half * 8;
            out[((size_t)(b * T_ + t)) * C_ + h * D_ + d] = acc[df][r] / lrow[r];
        }
    }
}

// ---------------------------------------------------------------------------
// out_bf = bf16(gamma * attn + x)
// ---------------------------------------------------------------------------
__global__ void residual_cvt_kernel(const float* __restrict__ attn, const float* __restrict__ x,
                                    const float* __restrict__ gamma, __bf16* __restrict__ out, int n) {
    int i = blockIdx.x * 256 + threadIdx.x;
    if (i < n) out[i] = (__bf16)(gamma[0] * attn[i] + x[i]);
}

// ---------------------------------------------------------------------------
// Per-row LayerNorm (P=1280) + cubic Bezier activation.
// ---------------------------------------------------------------------------
__global__ __launch_bounds__(256)
void ln_bezier_kernel(const float* __restrict__ hin, const float* __restrict__ g,
                      const float* __restrict__ bta, const float* __restrict__ bez,
                      float* __restrict__ out) {
    __shared__ float sbuf[8], ssbuf[8];
    int row = blockIdx.x;
    const float* hr = hin + (size_t)row * P_;
    int wave = threadIdx.x >> 5, lane = threadIdx.x & 31;

    float s = 0.0f, ss = 0.0f;
    for (int c = threadIdx.x; c < P_; c += 256) { float v = hr[c]; s += v; ss += v * v; }
#pragma unroll
    for (int off = 16; off >= 1; off >>= 1) { s += __shfl_xor(s, off, 32); ss += __shfl_xor(ss, off, 32); }
    if (lane == 0) { sbuf[wave] = s; ssbuf[wave] = ss; }
    __syncthreads();
    if (wave == 0) {
        float a  = (lane < 8) ? sbuf[lane]  : 0.0f;
        float a2 = (lane < 8) ? ssbuf[lane] : 0.0f;
#pragma unroll
        for (int off = 4; off >= 1; off >>= 1) { a += __shfl_xor(a, off, 32); a2 += __shfl_xor(a2, off, 32); }
        if (lane == 0) { sbuf[0] = a; ssbuf[0] = a2; }
    }
    __syncthreads();
    float mean = sbuf[0] * (1.0f / P_);
    float var  = ssbuf[0] * (1.0f / P_) - mean * mean;
    float inv  = rsqrtf(var + 1e-5f);
    float p0 = bez[0], p1 = bez[1], p2 = bez[2], p3 = bez[3];
    for (int c = threadIdx.x; c < P_; c += 256) {
        float z  = (hr[c] - mean) * inv * g[c] + bta[c];
        float sg = 1.0f / (1.0f + __expf(-z));
        float u  = 1.0f - sg;
        out[(size_t)row * P_ + c] =
            u * u * u * p0 + 3.0f * u * u * sg * p1 + 3.0f * u * sg * sg * p2 + sg * sg * sg * p3;
    }
}

// ---------------------------------------------------------------------------
// Host launcher
// ---------------------------------------------------------------------------
extern "C" void kernel_launch(void* const* d_in, const int* in_sizes, int n_in,
                              void* d_out, int out_size, void* d_ws, size_t ws_size,
                              hipStream_t stream) {
    const float* x     = (const float*)d_in[0];
    const float* Wq    = (const float*)d_in[1];
    const float* bq    = (const float*)d_in[2];
    const float* Wk    = (const float*)d_in[3];
    const float* bk    = (const float*)d_in[4];
    const float* Wv    = (const float*)d_in[5];
    const float* bv    = (const float*)d_in[6];
    const float* gamma = (const float*)d_in[7];
    const float* Wp    = (const float*)d_in[8];
    const float* bp    = (const float*)d_in[9];
    const float* ln_g  = (const float*)d_in[10];
    const float* ln_b  = (const float*)d_in[11];
    const float* bez   = (const float*)d_in[12];
    float* outp = (float*)d_out;

    // --- workspace partition (256B aligned) ---
    char* ws = (char*)d_ws;
    size_t off = 0;
    auto take = [&](size_t bytes) { char* p = ws + off; off = (off + bytes + 255) & ~(size_t)255; return p; };
    __bf16* xbf  = (__bf16*)take((size_t)M_ * C_ * 2);      // 8 MB
    __bf16* wqT  = (__bf16*)take((size_t)C_ * C_ * 2);      // 2 MB
    __bf16* wkT  = (__bf16*)take((size_t)C_ * C_ * 2);
    __bf16* wvT  = (__bf16*)take((size_t)C_ * C_ * 2);
    __bf16* wpT  = (__bf16*)take((size_t)P_ * C_ * 2);      // 2.5 MB
    __bf16* qbf  = (__bf16*)take((size_t)M_ * C_ * 2);      // 8 MB
    __bf16* kbf  = (__bf16*)take((size_t)M_ * C_ * 2);
    __bf16* vTbf = (__bf16*)take((size_t)M_ * C_ * 2);
    float*  attn = (float*) take((size_t)M_ * C_ * 4);      // 16 MB
    __bf16* obf  = (__bf16*)take((size_t)M_ * C_ * 2);      // 8 MB
    float*  hbuf = (float*) take((size_t)M_ * P_ * 4);      // 20 MB
    (void)ws_size; (void)in_sizes; (void)n_in; (void)out_size;

    const int nXC = M_ * C_;
    // 1. converts / transposes
    cvt_bf16_kernel<<<(nXC + 255) / 256, 256, 0, stream>>>(x, xbf, nXC);
    transpose_cvt_kernel<<<(C_ * C_ + 255) / 256, 256, 0, stream>>>(Wq, wqT, C_, C_);
    transpose_cvt_kernel<<<(C_ * C_ + 255) / 256, 256, 0, stream>>>(Wk, wkT, C_, C_);
    transpose_cvt_kernel<<<(C_ * C_ + 255) / 256, 256, 0, stream>>>(Wv, wvT, C_, C_);
    transpose_cvt_kernel<<<(C_ * P_ + 255) / 256, 256, 0, stream>>>(Wp, wpT, C_, P_);

    // 2. QKV projections (WMMA)
    dim3 gqkv(M_ / 128, C_ / 128);
    gemm_bf16_kernel<<<gqkv, 256, 0, stream>>>(xbf, wqT, bq, qbf, 0);
    gemm_bf16_kernel<<<gqkv, 256, 0, stream>>>(xbf, wkT, bk, kbf, 1);
    gemm_bf16_kernel<<<gqkv, 256, 0, stream>>>(xbf, wvT, bv, vTbf, 2);

    // 3. flash attention (WMMA)
    dim3 gattn(B_ * H_, T_ / 128);
    attn_kernel<<<gattn, 256, 0, stream>>>(qbf, kbf, vTbf, attn);

    // 4. gated residual -> bf16
    residual_cvt_kernel<<<(nXC + 255) / 256, 256, 0, stream>>>(attn, x, gamma, obf, nXC);

    // 5. output projection (WMMA)
    dim3 gproj(M_ / 128, P_ / 128);
    gemm_bf16_kernel<<<gproj, 256, 0, stream>>>(obf, wpT, bp, hbuf, 3);

    // 6. LayerNorm + Bezier
    ln_bezier_kernel<<<M_, 256, 0, stream>>>(hbuf, ln_g, ln_b, bez, outp);
}
